// VectorQuantizer_15496242004776
// MI455X (gfx1250) — matrix operand order repaired and checked by reading, not compiled, not run on hardware
//
#include <hip/hip_runtime.h>
#include <hip/hip_bf16.h>
#include <math.h>

// ---------------- problem constants (hardcoded from reference) ----------------
#define NB    32            // batch
#define DCH   256           // channels / embedding dim
#define HW    1024          // 32*32
#define NROWS 32768         // NB*HW
#define KCODE 1024          // codebook size
#define QELEMS 8388608      // NB*DCH*HW
// d_out float offsets
#define OFF_LOSS 0
#define OFF_Q    1
#define OFF_PERP (1 + QELEMS)
#define OFF_ENC  (2 + QELEMS)

typedef __attribute__((ext_vector_type(16))) __bf16 v16bf;
typedef __attribute__((ext_vector_type(8)))  float  v8f;

union Frag16 {
    v16bf v;
    unsigned short u[16];
    uint4 q[2];
};

static __device__ __forceinline__ unsigned short f32_to_bf16(float f) {
    unsigned int u = __float_as_uint(f);
    unsigned int lsb = (u >> 16) & 1u;
    u += 0x7fffu + lsb;               // round-to-nearest-even
    return (unsigned short)(u >> 16);
}

// ---------------------------------------------------------------------------
// Kernel 1: wnorm, bf16 copy of W, zero accumulators
// grid = KCODE blocks, 256 threads
// ---------------------------------------------------------------------------
__global__ __launch_bounds__(256) void prep_kernel(
    const float* __restrict__ W, float* __restrict__ wnorm,
    unsigned short* __restrict__ Wbf, float* __restrict__ probs,
    float* __restrict__ lsum)
{
    __shared__ float red[256];
    int k = blockIdx.x;
    int tid = threadIdx.x;
    float v = W[(size_t)k * DCH + tid];
    Wbf[(size_t)k * DCH + tid] = f32_to_bf16(v);
    red[tid] = v * v;
    __syncthreads();
    for (int s = 128; s > 0; s >>= 1) {
        if (tid < s) red[tid] += red[tid + s];
        __syncthreads();
    }
    if (tid == 0) { wnorm[k] = red[0]; probs[k] = 0.0f; }
    if (k == 0 && tid == 1) lsum[0] = 0.0f;
}

// ---------------------------------------------------------------------------
// Kernel 2: bf16 WMMA distance GEMM + fused top-2 + weights + avg_probs atomics
// M-tile = 32 rows/block (halves L2 codebook traffic vs 16):
//   each B fragment feeds 2 WMMAs against 2 resident A-fragment sets.
// grid = NROWS/32 = 1024 blocks, 256 threads (8 waves)
// ---------------------------------------------------------------------------
__global__ __launch_bounds__(256) void dist_top2_kernel(
    const float* __restrict__ x, const float* __restrict__ wnorm,
    const unsigned short* __restrict__ Wbf, float* __restrict__ probs,
    int* __restrict__ idx0, int* __restrict__ idx1,
    float* __restrict__ w0arr, float* __restrict__ w1arr)
{
    __shared__ unsigned short Abf[32 * DCH];   // 16 KB (32 rows x 256, bf16)
    __shared__ float dist[32 * 256];           // 32 KB (quarter of K columns per pass)
    __shared__ float xnp[256];
    __shared__ float xnorm[32];
    __shared__ float pd0[256], pd1[256];
    __shared__ int   pi0[256], pi1[256];

    const int tid  = threadIdx.x;
    const int row0 = blockIdx.x * 32;          // 32 rows share the same image b
    const int b    = row0 >> 10;
    const int hw0  = row0 & 1023;

    // ---- load X tile (NCHW gather), convert to bf16, partial row norms ----
    const int r     = tid & 31;                // row within tile (== hw offset)
    const int dbase = tid >> 5;                // 0..7
    const float* xb = x + (size_t)b * (DCH * HW) + hw0 + r;
    float sq = 0.0f;
#pragma unroll
    for (int it = 0; it < 32; ++it) {
        int d = dbase * 32 + it;
        float v = xb[(size_t)d * HW];
        sq += v * v;
        Abf[r * DCH + d] = f32_to_bf16(v);
    }
    xnp[tid] = sq;
    __syncthreads();
    if (tid < 32) {
        float s = 0.0f;
#pragma unroll
        for (int j = 0; j < 8; ++j) s += xnp[tid + 32 * j];
        xnorm[tid] = s;
    }
    __syncthreads();

    // ---- preload A fragments: 2 M-tiles x 8 K-chunks (held across all passes) ----
    const int wave = tid >> 5;
    const int lane = tid & 31;
    const int nloc = lane & 15;                // A: M-row / B: N-col / D: N-col
    const int half = lane >> 4;

    v16bf afrag[2][8];
#pragma unroll
    for (int mt = 0; mt < 2; ++mt) {
#pragma unroll
        for (int c = 0; c < 8; ++c) {
            Frag16 t;
#pragma unroll
            for (int e = 0; e < 16; ++e) {
                // A layout (matches B): half 0 -> K 0..15 ; half 1 -> K 16..31
                int k = e + 16 * half;
                t.u[e] = Abf[(mt * 16 + nloc) * DCH + c * 32 + k];
            }
            afrag[mt][c] = t.v;
        }
    }

    // ---- four passes over the 1024 codes (256 columns staged in LDS each) ----
    float d0 = 3.4e38f, d1 = 3.4e38f;
    int   i0 = -1,      i1 = -1;

    const int srow = tid >> 3;                 // scan role: row 0..31
    const int sseg = tid & 7;                  // scan role: 8 interleaved lanes

    for (int pass = 0; pass < 4; ++pass) {
        // GEMM role: this wave's 2 N-tiles in this pass
#pragma unroll
        for (int j = 0; j < 2; ++j) {
            const int t   = pass * 16 + wave + 8 * j;  // tile id 0..63
            const int n0  = t * 16;                    // global code base
            const int n0l = (t - pass * 16) * 16;      // column base inside dist[]
            v8f acc0 = {0.f, 0.f, 0.f, 0.f, 0.f, 0.f, 0.f, 0.f};
            v8f acc1 = {0.f, 0.f, 0.f, 0.f, 0.f, 0.f, 0.f, 0.f};
            // B layout: lane = n + 16*half ; elements e -> k = 16*half + e (contiguous)
            const unsigned short* wp =
                Wbf + (size_t)(n0 + nloc) * DCH + half * 16;
#pragma unroll
            for (int c = 0; c < 8; ++c) {
                Frag16 bt;
                const uint4* p = (const uint4*)(wp + c * 32);
                bt.q[0] = p[0];
                bt.q[1] = p[1];
                acc0 = __builtin_amdgcn_wmma_f32_16x16x32_bf16(
                    false, afrag[0][c], false, bt.v, (short)0, acc0, false, false);
                acc1 = __builtin_amdgcn_wmma_f32_16x16x32_bf16(
                    false, afrag[1][c], false, bt.v, (short)0, acc1, false, false);
            }
            const float wn = wnorm[n0 + nloc];
#pragma unroll
            for (int rr = 0; rr < 8; ++rr) {
                int m = rr + 8 * half;         // D layout: vgpr rr -> M, lane%16 -> N
                dist[m * 256 + n0l + nloc]        = xnorm[m]      + wn - 2.0f * acc0[rr];
                dist[(16 + m) * 256 + n0l + nloc] = xnorm[16 + m] + wn - 2.0f * acc1[rr];
            }
        }
        __syncthreads();

        // scan role: 8 threads/row, interleaved columns (ascending per thread)
        const float* dr = dist + srow * 256;
#pragma unroll 4
        for (int j = 0; j < 32; ++j) {
            int cl = sseg + 8 * j;
            float v = dr[cl];
            int ci = pass * 256 + cl;
            if (v < d0)      { d1 = d0; i1 = i0; d0 = v; i0 = ci; }
            else if (v < d1) { d1 = v; i1 = ci; }
        }
        __syncthreads();                       // dist[] reused next pass
    }

    pd0[tid] = d0; pi0[tid] = i0; pd1[tid] = d1; pi1[tid] = i1;
    __syncthreads();

    if (tid < 32) {
        float b0 = 3.4e38f, b1 = 3.4e38f;
        int   j0 = -1,      j1 = -1;
        for (int s = 0; s < 8; ++s) {
            int p = tid * 8 + s;
            float v = pd0[p]; int ci = pi0[p];
            if (v < b0)      { b1 = b0; j1 = j0; b0 = v; j0 = ci; }
            else if (v < b1) { b1 = v; j1 = ci; }
            v = pd1[p]; ci = pi1[p];
            if (v < b0)      { b1 = b0; j1 = j0; b0 = v; j0 = ci; }
            else if (v < b1) { b1 = v; j1 = ci; }
        }
        float inv0 = 1.0f / b0, inv1 = 1.0f / b1;
        float nrm = fmaxf(sqrtf(inv0 * inv0 + inv1 * inv1), 1e-12f);
        float w0 = inv0 / nrm, w1 = inv1 / nrm;
        int grow = row0 + tid;
        idx0[grow] = j0; idx1[grow] = j1;
        w0arr[grow] = w0; w1arr[grow] = w1;
        atomicAdd(&probs[j0], w0);
        atomicAdd(&probs[j1], w1);
    }
}

// ---------------------------------------------------------------------------
// Kernel 3: scatter encodings [NROWS, KCODE] (float2 stores; base is 8B aligned)
// grid = NROWS blocks, 256 threads
// ---------------------------------------------------------------------------
__global__ __launch_bounds__(256) void encodings_kernel(
    const int* __restrict__ idx0, const int* __restrict__ idx1,
    const float* __restrict__ w0a, const float* __restrict__ w1a,
    float* __restrict__ enc)
{
    int row = blockIdx.x;
    int i0 = idx0[row], i1 = idx1[row];
    float w0 = w0a[row], w1 = w1a[row];
    float2* out = (float2*)(enc + (size_t)row * KCODE);
#pragma unroll
    for (int it = 0; it < 2; ++it) {
        int p = it * 256 + threadIdx.x;        // float2 index, k = 2p, 2p+1
        int k = p * 2;
        float2 v;
        v.x = (k     == i0) ? w0 : ((k     == i1) ? w1 : 0.0f);
        v.y = (k + 1 == i0) ? w0 : ((k + 1 == i1) ? w1 : 0.0f);
        out[p] = v;
    }
}

// ---------------------------------------------------------------------------
// Kernel 4: quantized_st (NCHW, via LDS transpose) + fused sum of (q-x)^2
// grid = HW blocks (1024), 256 threads; each block: 32 rows x 256 channels
// ---------------------------------------------------------------------------
__global__ __launch_bounds__(256) void quantized_kernel(
    const float* __restrict__ x, const float* __restrict__ W,
    const int* __restrict__ idx0, const int* __restrict__ idx1,
    const float* __restrict__ w0a, const float* __restrict__ w1a,
    float* __restrict__ outq, float* __restrict__ lsum)
{
    __shared__ float q[256 * 33];              // [d][r], padded: conflict-free
    __shared__ float red[256];
    const int tid  = threadIdx.x;              // = d in stage 1
    const int row0 = blockIdx.x * 32;
    const int b    = row0 >> 10;
    const int hw0  = row0 & 1023;

    for (int r = 0; r < 32; ++r) {
        int grow = row0 + r;
        int i0 = idx0[grow], i1 = idx1[grow];
        float w0 = w0a[grow], w1 = w1a[grow];
        float qv = w0 * W[(size_t)i0 * DCH + tid] + w1 * W[(size_t)i1 * DCH + tid];
        q[tid * 33 + r] = qv;
    }
    __syncthreads();

    float acc = 0.0f;
    const float* xb = x    + (size_t)b * (DCH * HW) + hw0;
    float*       ob = outq + (size_t)b * (DCH * HW) + hw0;
#pragma unroll 4
    for (int it = 0; it < 32; ++it) {
        int rr = tid & 31;
        int d  = it * 8 + (tid >> 5);
        float qv = q[d * 33 + rr];
        size_t off = (size_t)d * HW + rr;
        float xv = xb[off];
        float diff = qv - xv;
        ob[off] = xv + diff;                   // straight-through: x + (q - x)
        acc += diff * diff;
    }
    red[tid] = acc;
    __syncthreads();
    for (int s = 128; s > 0; s >>= 1) {
        if (tid < s) red[tid] += red[tid + s];
        __syncthreads();
    }
    if (tid == 0) atomicAdd(lsum, red[0]);
}

// ---------------------------------------------------------------------------
// Kernel 5: loss + perplexity scalars
// ---------------------------------------------------------------------------
__global__ __launch_bounds__(256) void finalize_kernel(
    const float* __restrict__ probs, const float* __restrict__ lsum,
    float* __restrict__ out)
{
    __shared__ float red[256];
    int tid = threadIdx.x;
    float s = 0.0f;
#pragma unroll
    for (int j = 0; j < 4; ++j) {
        float p = probs[tid * 4 + j] * (1.0f / (float)NROWS);
        s += -p * logf(p + 1e-10f);
    }
    red[tid] = s;
    __syncthreads();
    for (int st = 128; st > 0; st >>= 1) {
        if (tid < st) red[tid] += red[tid + st];
        __syncthreads();
    }
    if (tid == 0) {
        out[OFF_PERP] = expf(red[0]);
        out[OFF_LOSS] = 1.25f * lsum[0] / (float)QELEMS;   // q_loss + 0.25*e_loss
    }
}

// ---------------------------------------------------------------------------
extern "C" void kernel_launch(void* const* d_in, const int* in_sizes, int n_in,
                              void* d_out, int out_size, void* d_ws, size_t ws_size,
                              hipStream_t stream) {
    (void)in_sizes; (void)n_in; (void)out_size; (void)ws_size;
    const float* x = (const float*)d_in[0];
    const float* W = (const float*)d_in[1];
    float* out = (float*)d_out;

    // workspace layout (float offsets)
    float* ws     = (float*)d_ws;
    float* wnorm  = ws;                         // 1024
    float* probs  = ws + 1024;                  // 1024
    float* lsum   = ws + 2048;                  // 1 (+pad to 2064)
    int*   idx0   = (int*)(ws + 2064);          // 32768
    int*   idx1   = (int*)(ws + 34832);         // 32768
    float* w0a    = ws + 67600;                 // 32768
    float* w1a    = ws + 100368;                // 32768
    unsigned short* Wbf = (unsigned short*)(ws + 133136); // 262144 bf16 (16B-aligned)

    prep_kernel<<<KCODE, 256, 0, stream>>>(W, wnorm, Wbf, probs, lsum);
    dist_top2_kernel<<<NROWS / 32, 256, 0, stream>>>(
        x, wnorm, Wbf, probs, idx0, idx1, w0a, w1a);
    encodings_kernel<<<NROWS, 256, 0, stream>>>(
        idx0, idx1, w0a, w1a, out + OFF_ENC);
    quantized_kernel<<<HW, 256, 0, stream>>>(
        x, W, idx0, idx1, w0a, w1a, out + OFF_Q, lsum);
    finalize_kernel<<<1, 256, 0, stream>>>(probs, lsum, out);
}
